// RoPEAttention_10033043603797
// MI455X (gfx1250) — compile-verified, matrix operand order checked by reading, not run on hardware
//
#include <hip/hip_runtime.h>
#include <hip/hip_bf16.h>
#include <math.h>

// ---------------------------------------------------------------------------
// RoPE causal attention, bf16 WMMA pipeline for gfx1250 (MI455X).
// B=2, T=2048, d_model=2048, H=16, Dh=128 (fixed by the reference).
// ---------------------------------------------------------------------------

typedef __attribute__((ext_vector_type(16))) __bf16 v16bf;
typedef __attribute__((ext_vector_type(8)))  float  v8f;
typedef __attribute__((ext_vector_type(4)))  int    v4i;

union TileU { uint4 u[2]; v16bf v; };

#if __has_builtin(__builtin_amdgcn_global_load_async_to_lds_b128) && \
    __has_builtin(__builtin_amdgcn_s_wait_asynccnt)
#define HAVE_ASYNC_LDS 1
#else
#define HAVE_ASYNC_LDS 0
#endif

__device__ __forceinline__ unsigned short f32_to_bf16(float f) {
  unsigned u = __float_as_uint(f);
  return (unsigned short)((u + 0x7FFFu + ((u >> 16) & 1u)) >> 16);
}
__device__ __forceinline__ float bf16_to_f32(unsigned short h) {
  return __uint_as_float(((unsigned)h) << 16);
}

// Copy 16 bytes global -> LDS via the gfx1250 async path when available.
// Builtin signature (from compiler diagnostic): param0 = v4i __device__*
// (addrspace(1) global source), then LDS dest, imm offset, imm cpol.
__device__ __forceinline__ void copy16_g2l(const unsigned short* gsrc,
                                           unsigned short* ldst) {
#if HAVE_ASYNC_LDS
  __builtin_amdgcn_global_load_async_to_lds_b128(
      (__attribute__((address_space(1))) v4i*)gsrc,
      (__attribute__((address_space(3))) v4i*)ldst, 0, 0);
#else
  *(uint4*)ldst = *(const uint4*)gsrc;
#endif
}
__device__ __forceinline__ void async_wait_all() {
#if HAVE_ASYNC_LDS
  __builtin_amdgcn_s_wait_asynccnt(0);
#endif
}

// Load one 16-element-per-lane operand chunk matching the 16-bit A/B VGPR
// layout: this lane's row, K elements {base..base+7, base+16..base+23}.
// Caller passes p = rowptr + kk + (lane>=16 ? 8 : 0).
__device__ __forceinline__ v16bf load_tile16x32(const unsigned short* p) {
  TileU t;
  t.u[0] = *(const uint4*)(p);
  t.u[1] = *(const uint4*)(p + 16);
  return t.v;
}

__device__ __forceinline__ v8f wmma_bf16(v16bf a, v16bf b, v8f c) {
  // (neg_a, A, neg_b, B, c_mod, C, reuse_a, reuse_b)
  return __builtin_amdgcn_wmma_f32_16x16x32_bf16(
      false, a, false, b, (short)0, c, false, false);
}

// ---------------------------------------------------------------------------
// f32 -> bf16 elementwise convert
// ---------------------------------------------------------------------------
__global__ __launch_bounds__(256) void k_cvt_bf16(
    const float* __restrict__ in, unsigned short* __restrict__ out, int n) {
  int i = blockIdx.x * blockDim.x + threadIdx.x;
  if (i < n) out[i] = f32_to_bf16(in[i]);
}

// ---------------------------------------------------------------------------
// C(M,N) = A(M,K) @ B(N,K)^T   (A,B bf16, accumulate f32)
// Block = 4 waves; block tile 64 rows x 128 cols; wave tile 16x128.
// B tile (128 cols x 32 K) staged in LDS via async copies, shared by waves.
// ---------------------------------------------------------------------------
template <bool F32OUT>
__global__ __launch_bounds__(128, 1) void k_gemm_bf16(
    const unsigned short* __restrict__ A, const unsigned short* __restrict__ Bm,
    void* __restrict__ Cp, int M, int N, int K) {
  __shared__ unsigned short ldsB[128 * 32];  // 8 KB: 128 cols x 32 K-chunk

  const int tid = threadIdx.x;
  const int lane = tid & 31;
  const int wave = tid >> 5;
  const int n0 = blockIdx.x * 128;
  const int m0 = blockIdx.y * 64 + wave * 16;
  const int nl = lane & 15;
  const int hh = lane >> 4;  // which 16-lane half (K-pattern select)

  const unsigned short* arow = A + (size_t)(m0 + nl) * K + hh * 8;
  const unsigned short* brow = Bm + (size_t)(n0 + tid) * K;  // this thread's col
  v8f acc[8] = {};

  for (int kk = 0; kk < K; kk += 32) {
    // Stage B[n0..n0+127][kk..kk+31] into LDS (each thread copies one row)
    copy16_g2l(brow + kk + 0, &ldsB[tid * 32 + 0]);
    copy16_g2l(brow + kk + 8, &ldsB[tid * 32 + 8]);
    copy16_g2l(brow + kk + 16, &ldsB[tid * 32 + 16]);
    copy16_g2l(brow + kk + 24, &ldsB[tid * 32 + 24]);
    async_wait_all();
    __syncthreads();

    v16bf a = load_tile16x32(arow + kk);
    if (kk + 64 < K) __builtin_prefetch(arow + kk + 64, 0, 1);
#pragma unroll
    for (int j = 0; j < 8; ++j) {
      v16bf b = load_tile16x32(&ldsB[(j * 16 + nl) * 32 + hh * 8]);
      acc[j] = wmma_bf16(a, b, acc[j]);
    }
    __syncthreads();  // before LDS tile is overwritten
  }

  // C layout: VGPR r holds M = r (lanes 0-15) / r+8 (lanes 16-31), N = lane&15
#pragma unroll
  for (int j = 0; j < 8; ++j) {
#pragma unroll
    for (int r = 0; r < 8; ++r) {
      int row = m0 + r + hh * 8;
      int col = n0 + j * 16 + nl;
      float v = acc[j][r];
      if (F32OUT)
        ((float*)Cp)[(size_t)row * N + col] = v;
      else
        ((unsigned short*)Cp)[(size_t)row * N + col] = f32_to_bf16(v);
    }
  }
}

// ---------------------------------------------------------------------------
// In-place RoPE on bf16 (B,T,H,Dh) tensor; `scale` folds in 1/sqrt(Dh) for q.
// One thread per (b,t,h,pair), pair fastest: i = ((b*T+t)*16+h)*64 + p.
// ---------------------------------------------------------------------------
__global__ __launch_bounds__(256) void k_rope(unsigned short* __restrict__ X,
                                              int T, int total, float scale) {
  int i = blockIdx.x * blockDim.x + threadIdx.x;
  if (i >= total) return;
  int p = i & 63;
  int t = (i >> 10) % T;  // (i / (64*16)) % T
  // theta = BASE^(-2p/128) = exp(-2p/128 * ln(10000))
  float theta = __expf(-(float)(2 * p) * (9.210340371976184f / 128.0f));
  float ang = (float)t * theta;
  float s, c;
  __sincosf(ang, &s, &c);
  size_t base = ((size_t)(i >> 6) << 7) + 2 * p;
  float x0 = bf16_to_f32(X[base]);
  float x1 = bf16_to_f32(X[base + 1]);
  X[base]     = f32_to_bf16((x0 * c - x1 * s) * scale);
  X[base + 1] = f32_to_bf16((x0 * s + x1 * c) * scale);
}

// ---------------------------------------------------------------------------
// V (B,T,H*Dh) -> Vt (B,H,Dh,T) so the PV B-operand is contiguous along keys.
// ---------------------------------------------------------------------------
__global__ __launch_bounds__(256) void k_transpose_v(
    const unsigned short* __restrict__ V, unsigned short* __restrict__ Vt,
    int T, int total) {
  int i = blockIdx.x * blockDim.x + threadIdx.x;
  if (i >= total) return;
  int d = i & 127;
  int h = (i >> 7) & 15;
  int t = (i >> 11) % T;
  int b = i / (T << 11);
  Vt[((size_t)(b * 16 + h) * 128 + d) * T + t] = V[i];
}

// ---------------------------------------------------------------------------
// Flash-attention core: one wave per (b,h, 16-row q tile); 32-key tiles.
// Q,K: (B,T,2048) bf16 (q pre-scaled by 1/sqrt(Dh)); Vt: (B,H,128,T) bf16.
// O: (B,T,2048) bf16.
// ---------------------------------------------------------------------------
__global__ __launch_bounds__(32, 1) void k_attn(
    const unsigned short* __restrict__ Q, const unsigned short* __restrict__ Kp,
    const unsigned short* __restrict__ Vt, unsigned short* __restrict__ O,
    int T) {
  __shared__ unsigned short ldsP[16 * 32];  // P tile staging (C -> A layout)
  const int lane = threadIdx.x & 31;
  const int nl = lane & 15;
  const int hh = lane >> 4;
  const int qbase = blockIdx.x * 16;
  const int bh = blockIdx.y;  // b*16 + h
  const int b = bh >> 4, h = bh & 15;

  const unsigned short* qP = Q + (size_t)(b * T) * 2048 + h * 128;
  const unsigned short* kP = Kp + (size_t)(b * T) * 2048 + h * 128;
  const unsigned short* vP = Vt + (size_t)bh * 128 * T;

  // Q A-operands for the 4 K-chunks of d_head=128, loaded once.
  v16bf qA[4];
#pragma unroll
  for (int c = 0; c < 4; ++c)
    qA[c] = load_tile16x32(qP + (size_t)(qbase + nl) * 2048 + c * 32 + hh * 8);

  v8f oacc[8] = {};
  float mrow[8], lrow[8];
#pragma unroll
  for (int r = 0; r < 8; ++r) { mrow[r] = -1e30f; lrow[r] = 0.f; }

  for (int jb = 0; jb < qbase + 16; jb += 32) {
    // S = Q K^T for 32 keys: two 16x16 tiles
    v8f s0 = {}, s1 = {};
#pragma unroll
    for (int c = 0; c < 4; ++c) {
      v16bf b0 =
          load_tile16x32(kP + (size_t)(jb + nl) * 2048 + c * 32 + hh * 8);
      v16bf b1 =
          load_tile16x32(kP + (size_t)(jb + 16 + nl) * 2048 + c * 32 + hh * 8);
      s0 = wmma_bf16(qA[c], b0, s0);
      s1 = wmma_bf16(qA[c], b1, s1);
    }

    // Causal mask + online softmax. Row m = r + 8*hh, col key = jb+nl (+16).
    float alpha[8];
#pragma unroll
    for (int r = 0; r < 8; ++r) {
      int qrow = qbase + r + hh * 8;
      float v0 = (jb + nl <= qrow) ? s0[r] : -1e30f;
      float v1 = (jb + 16 + nl <= qrow) ? s1[r] : -1e30f;
      float mx = fmaxf(v0, v1);
      mx = fmaxf(mx, __shfl_xor(mx, 1));
      mx = fmaxf(mx, __shfl_xor(mx, 2));
      mx = fmaxf(mx, __shfl_xor(mx, 4));
      mx = fmaxf(mx, __shfl_xor(mx, 8));
      float mnew = fmaxf(mrow[r], mx);
      float a = __expf(mrow[r] - mnew);
      float p0 = __expf(v0 - mnew);
      float p1 = __expf(v1 - mnew);
      float sum = p0 + p1;
      sum += __shfl_xor(sum, 1);
      sum += __shfl_xor(sum, 2);
      sum += __shfl_xor(sum, 4);
      sum += __shfl_xor(sum, 8);
      lrow[r] = lrow[r] * a + sum;
      mrow[r] = mnew;
      alpha[r] = a;
      int m = r + hh * 8;
      ldsP[m * 32 + nl] = f32_to_bf16(p0);
      ldsP[m * 32 + 16 + nl] = f32_to_bf16(p1);
    }

    // Rescale running output by per-row alpha
#pragma unroll
    for (int d = 0; d < 8; ++d)
#pragma unroll
      for (int r = 0; r < 8; ++r) oacc[d][r] *= alpha[r];

    // Reload P in A-operand layout (row = lane&15) from LDS; single-wave,
    // DS ops are in-order per wave.
    v16bf pA = load_tile16x32(&ldsP[nl * 32 + hh * 8]);

    // O += P @ V : B-operand lane = d-column, contiguous along keys in Vt.
#pragma unroll
    for (int d = 0; d < 8; ++d) {
      v16bf bv = load_tile16x32(vP + (size_t)(d * 16 + nl) * T + jb + hh * 8);
      oacc[d] = wmma_bf16(pA, bv, oacc[d]);
    }
  }

  // Normalize and store O (bf16)
#pragma unroll
  for (int d = 0; d < 8; ++d)
#pragma unroll
    for (int r = 0; r < 8; ++r) {
      int row = qbase + r + hh * 8;
      float v = oacc[d][r] / lrow[r];
      O[(size_t)(b * T + row) * 2048 + h * 128 + d * 16 + nl] = f32_to_bf16(v);
    }
}

// ---------------------------------------------------------------------------
// Host launcher
// ---------------------------------------------------------------------------
extern "C" void kernel_launch(void* const* d_in, const int* in_sizes, int n_in,
                              void* d_out, int out_size, void* d_ws,
                              size_t ws_size, hipStream_t stream) {
  (void)n_in; (void)out_size; (void)ws_size; (void)in_sizes;
  const int B = 2, T = 2048, D = 2048;
  const size_t NB = (size_t)B * T * D;  // activation elements
  const size_t WB = (size_t)D * D;      // weight elements

  const float* x = (const float*)d_in[0];
  const float* wq = (const float*)d_in[1];
  const float* wk = (const float*)d_in[2];
  const float* wv = (const float*)d_in[3];
  const float* wo = (const float*)d_in[4];
  float* out = (float*)d_out;

  unsigned short* ws = (unsigned short*)d_ws;
  unsigned short* xb = ws;
  unsigned short* wqb = xb + NB;
  unsigned short* wkb = wqb + WB;
  unsigned short* wvb = wkb + WB;
  unsigned short* wob = wvb + WB;
  unsigned short* qb = wob + WB;
  unsigned short* kb = qb + NB;
  unsigned short* vb = kb + NB;
  unsigned short* vtb = vb + NB;
  unsigned short* ob = vtb + NB;

  // 1) f32 -> bf16 conversions
  {
    int th = 256;
    k_cvt_bf16<<<(int)((NB + th - 1) / th), th, 0, stream>>>(x, xb, (int)NB);
    k_cvt_bf16<<<(int)((WB + th - 1) / th), th, 0, stream>>>(wq, wqb, (int)WB);
    k_cvt_bf16<<<(int)((WB + th - 1) / th), th, 0, stream>>>(wk, wkb, (int)WB);
    k_cvt_bf16<<<(int)((WB + th - 1) / th), th, 0, stream>>>(wv, wvb, (int)WB);
    k_cvt_bf16<<<(int)((WB + th - 1) / th), th, 0, stream>>>(wo, wob, (int)WB);
  }

  // 2) QKV projections (bf16 out)
  {
    dim3 grid(D / 128, (B * T) / 64);
    dim3 block(128);
    k_gemm_bf16<false><<<grid, block, 0, stream>>>(xb, wqb, qb, B * T, D, D);
    k_gemm_bf16<false><<<grid, block, 0, stream>>>(xb, wkb, kb, B * T, D, D);
    k_gemm_bf16<false><<<grid, block, 0, stream>>>(xb, wvb, vb, B * T, D, D);
  }

  // 3) RoPE on q (with 1/sqrt(Dh) folded in) and k
  {
    int total = B * T * 16 * 64;
    int th = 256;
    k_rope<<<(total + th - 1) / th, th, 0, stream>>>(qb, T, total,
                                                     0.08838834764831845f);
    k_rope<<<(total + th - 1) / th, th, 0, stream>>>(kb, T, total, 1.0f);
  }

  // 4) V -> Vt (B,H,Dh,T)
  {
    int th = 256;
    k_transpose_v<<<(int)((NB + th - 1) / th), th, 0, stream>>>(vb, vtb, T,
                                                                (int)NB);
  }

  // 5) Flash attention core
  {
    dim3 grid(T / 16, B * 16);
    k_attn<<<grid, dim3(32), 0, stream>>>(qb, kb, vtb, ob, T);
  }

  // 6) Output projection (f32 out)
  {
    dim3 grid(D / 128, (B * T) / 64);
    dim3 block(128);
    k_gemm_bf16<true><<<grid, block, 0, stream>>>(ob, wob, out, B * T, D, D);
  }
}
